// Decoding_77841987272844
// MI455X (gfx1250) — matrix operand order repaired and checked by reading, not compiled, not run on hardware
//
#include <hip/hip_runtime.h>
#include <hip/hip_bf16.h>
#include <math.h>
#include <stdint.h>

#define NBINS 500
#define NCELLS 128
#define NGOI 300
#define NHID 32
#define NLAT 64
#define NFRAG 50000
#define NSPL 1499
#define WINA (-10000.0f)
#define WINB (10000.0f)
#define MINW 0.001f
#define MINH 0.001f
#define MIND 0.001f

// bank-conflict-free padded LDS index: stride-17 chunks over 64 banks
#define PAD(j) ((j) + ((j) >> 4))
#define PADDED 544   // PAD(511)=542, round up

typedef __attribute__((ext_vector_type(16))) _Float16 v16h;
typedef __attribute__((ext_vector_type(8)))  float    v8f;

// ---------- wave32 helpers ----------
__device__ __forceinline__ float wred_sum(float v) {
  for (int o = 16; o > 0; o >>= 1) v += __shfl_xor(v, o, 32);
  return v;
}
__device__ __forceinline__ float wred_max(float v) {
  for (int o = 16; o > 0; o >>= 1) v = fmaxf(v, __shfl_xor(v, o, 32));
  return v;
}
__device__ __forceinline__ int wred_sumi(int v) {
  for (int o = 16; o > 0; o >>= 1) v += __shfl_xor(v, o, 32);
  return v;
}
// exclusive scan across the 32 lanes
__device__ __forceinline__ float wscan_excl(float v, int lane) {
  float x = v;
  for (int o = 1; o < 32; o <<= 1) {
    float t = __shfl_up(x, o, 32);
    if (lane >= o) x += t;
  }
  return x - v;
}
__device__ __forceinline__ void lds_fence() {
  asm volatile("s_wait_dscnt 0x0" ::: "memory");
}
// uniform broadcast of lane k's value (k literal after unroll): v_readlane -> SGPR
__device__ __forceinline__ float bcast(float v, int k) {
  return __builtin_bit_cast(float,
           __builtin_amdgcn_readlane(__builtin_bit_cast(int, v), k));
}

// ---------- kernel 0: zero scalar output + counts scratch ----------
__global__ void init_kernel(float* out, float* counts) {
  int i = blockIdx.x * blockDim.x + threadIdx.x;
  if (i == 0) out[0] = 0.0f;
  if (i < NCELLS * NGOI) counts[i] = 0.0f;
}

// ---------- kernel 1: MLP (2 layers, relu + batchnorm) with f16 WMMA ----------
// one block of 256 threads (8 waves). wave w owns rows [16w, 16w+16).
// LDS staging of inputs uses gfx1250 async global->LDS copies (ASYNCcnt).
__global__ void mlp_kernel(const float* __restrict__ latent,
                           const float* __restrict__ W1, const float* __restrict__ b1,
                           const float* __restrict__ g1, const float* __restrict__ be1,
                           const float* __restrict__ W2, const float* __restrict__ b2,
                           const float* __restrict__ g2, const float* __restrict__ be2,
                           float* __restrict__ hout) {
  __shared__ __align__(16) float Xs[NCELLS * NLAT];   // 32 KB (reused as layer-2 input)
  __shared__ __align__(16) float Ws[NLAT * NHID];     // 8 KB
  __shared__ __align__(16) float Ys[NCELLS * NHID];   // 16 KB
  __shared__ float mean_[NHID], scale_[NHID], beta_[NHID];
  const int t = threadIdx.x, lane = t & 31, wv = t >> 5;
  const int khalf = lane >> 4, mloc = lane & 15;

  // async DMA: global -> LDS, 16 B per lane per iteration, exact multiples of 256
  for (int i = t; i < (NCELLS * NLAT) / 4; i += 256) {    // 8 full iterations
    unsigned lds = (unsigned)(uintptr_t)(&Xs[i * 4]);
    const float* gp = latent + i * 4;
    asm volatile("global_load_async_to_lds_b128 %0, %1, off"
                 :: "v"(lds), "v"(gp) : "memory");
  }
  for (int i = t; i < (NLAT * NHID) / 4; i += 256) {      // 2 full iterations
    unsigned lds = (unsigned)(uintptr_t)(&Ws[i * 4]);
    const float* gp = W1 + i * 4;
    asm volatile("global_load_async_to_lds_b128 %0, %1, off"
                 :: "v"(lds), "v"(gp) : "memory");
  }
  asm volatile("s_wait_asynccnt 0x0" ::: "memory");
  __syncthreads();

  // ---- layer 1: [128x64] @ [64x32], K = 2 WMMA steps ----
  {
    const int m = wv * 16 + mloc;
    for (int nt = 0; nt < 2; ++nt) {
      const int n = nt * 16 + mloc;
      v8f cacc = {};
      for (int kc = 0; kc < 2; ++kc) {
        v16h a, b;
        for (int v = 0; v < 8; ++v) {
          int kb = kc * 32 + ((v < 4) ? 0 : 16) + khalf * 8 + (v & 3) * 2;
          a[2 * v]     = (_Float16)Xs[m * NLAT + kb];
          a[2 * v + 1] = (_Float16)Xs[m * NLAT + kb + 1];
        }
        for (int v = 0; v < 8; ++v) {
          int kb = kc * 32 + ((v < 4) ? 0 : 16) + khalf * 8 + (v & 3) * 2;
          b[2 * v]     = (_Float16)Ws[kb * NHID + n];
          b[2 * v + 1] = (_Float16)Ws[(kb + 1) * NHID + n];
        }
        cacc = __builtin_amdgcn_wmma_f32_16x16x32_f16(false, a, false, b,
                                                      (short)0, cacc, false, false);
      }
      for (int r = 0; r < 8; ++r) {
        int mm = wv * 16 + r + khalf * 8;
        Ys[mm * NHID + n] = fmaxf(cacc[r] + b1[n], 0.0f);
      }
    }
  }
  __syncthreads();
  if (t < NHID) {                       // batchnorm-1 column stats
    float s = 0.f, ss = 0.f;
    for (int r = 0; r < NCELLS; ++r) { float v = Ys[r * NHID + t]; s += v; ss += v * v; }
    float mn = s / NCELLS, var = ss / NCELLS - mn * mn;
    mean_[t] = mn; scale_[t] = g1[t] * rsqrtf(var + 1e-5f); beta_[t] = be1[t];
  }
  __syncthreads();
  for (int i = t; i < NCELLS * NHID; i += 256) {
    int col = i & 31;
    Xs[i] = (Ys[i] - mean_[col]) * scale_[col] + beta_[col];
  }
  for (int i = t; i < (NHID * NHID) / 4; i += 256) {      // 1 full iteration
    unsigned lds = (unsigned)(uintptr_t)(&Ws[i * 4]);
    const float* gp = W2 + i * 4;
    asm volatile("global_load_async_to_lds_b128 %0, %1, off"
                 :: "v"(lds), "v"(gp) : "memory");
  }
  asm volatile("s_wait_asynccnt 0x0" ::: "memory");
  __syncthreads();

  // ---- layer 2: [128x32] @ [32x32], single WMMA per tile ----
  {
    const int m = wv * 16 + mloc;
    for (int nt = 0; nt < 2; ++nt) {
      const int n = nt * 16 + mloc;
      v16h a, b; v8f cacc = {};
      for (int v = 0; v < 8; ++v) {
        int kb = ((v < 4) ? 0 : 16) + khalf * 8 + (v & 3) * 2;
        a[2 * v]     = (_Float16)Xs[m * NHID + kb];
        a[2 * v + 1] = (_Float16)Xs[m * NHID + kb + 1];
      }
      for (int v = 0; v < 8; ++v) {
        int kb = ((v < 4) ? 0 : 16) + khalf * 8 + (v & 3) * 2;
        b[2 * v]     = (_Float16)Ws[kb * NHID + n];
        b[2 * v + 1] = (_Float16)Ws[(kb + 1) * NHID + n];
      }
      cacc = __builtin_amdgcn_wmma_f32_16x16x32_f16(false, a, false, b,
                                                    (short)0, cacc, false, false);
      for (int r = 0; r < 8; ++r) {
        int mm = wv * 16 + r + khalf * 8;
        Ys[mm * NHID + n] = fmaxf(cacc[r] + b2[n], 0.0f);
      }
    }
  }
  __syncthreads();
  if (t < NHID) {                       // batchnorm-2 column stats
    float s = 0.f, ss = 0.f;
    for (int r = 0; r < NCELLS; ++r) { float v = Ys[r * NHID + t]; s += v; ss += v * v; }
    float mn = s / NCELLS, var = ss / NCELLS - mn * mn;
    mean_[t] = mn; scale_[t] = g2[t] * rsqrtf(var + 1e-5f); beta_[t] = be2[t];
  }
  __syncthreads();
  for (int i = t; i < NCELLS * NHID; i += 256) {
    int col = i & 31;
    hout[i] = (Ys[i] - mean_[col]) * scale_[col] + beta_[col];
  }
}

// ---------- kernel 2: histogram of cellxgene indices ----------
__global__ void hist_kernel(const int* __restrict__ lcxg, float* __restrict__ counts) {
  int i = blockIdx.x * blockDim.x + threadIdx.x;
  if (i < NFRAG) atomicAdd(&counts[lcxg[i]], 1.0f);
}

// on-demand derivative param: d = MIND + softplus(mix + h . sw[:,col])
__device__ __forceinline__ float dval(const float* __restrict__ swg,
                                      const float* __restrict__ mixg,
                                      float hv, int lane, int col) {
  float p = hv * swg[lane * NSPL + col];
  p = wred_sum(p);
  p += mixg[col];
  float sp = (p > 20.0f) ? p : log1pf(expf(p));
  return MIND + sp;
}

// ---------- kernel 3: fused fragment spline inverse log-det (one wave per fragment) ----------
__global__ void frag_kernel(const float* __restrict__ h, const float* __restrict__ coords,
                            const float* __restrict__ spline_w,
                            const float* __restrict__ mix_spline,
                            const int* __restrict__ genes_oi, const int* __restrict__ lcxg,
                            float* __restrict__ out) {
  __shared__ float ubuf[8][2][PADDED];   // ~34 KB: per-wave uw / uh staging (padded)
  const int lane = threadIdx.x & 31, wv = threadIdx.x >> 5;
  const int fid = blockIdx.x * 8 + wv;
  if (fid >= NFRAG) return;
  float* wbuf = ubuf[wv][0];
  float* hbuf = ubuf[wv][1];

  const int cxg = lcxg[fid];
  const int c = cxg / NGOI;
  const int gl = cxg - c * NGOI;
  const int g = genes_oi[gl];
  const float hv = h[c * NHID + lane];
  const float* swg = spline_w + (size_t)g * NHID * NSPL;
  const float* mixg = mix_spline + (size_t)g * NSPL;

  // warm L2/L0 for the mix row used across passes A and C
  __builtin_prefetch(mixg + lane * 47, 0, 3);

  // pass A: compute raw params (coalesced over bins), stage to LDS.
  // broadcast of the wave-uniform h row via v_readlane (SGPR operand FMAs).
  for (int i = 0; i < 16; ++i) {
    int j = i * 32 + lane;
    float pw = -1e30f, ph = -1e30f;
    if (j < NBINS) {
      pw = mixg[j]; ph = mixg[NBINS + j];
#pragma unroll
      for (int k = 0; k < 32; ++k) {
        float hk = bcast(hv, k);
        const float* row = swg + k * NSPL;
        pw += hk * row[j];
        ph += hk * row[NBINS + j];
      }
    }
    wbuf[PAD(j)] = pw; hbuf[PAD(j)] = ph;
  }
  lds_fence();

  // pass B: softmax + widths/heights + prefix sums (contiguous chunk per lane)
  const int base = lane * 16;
  const int pbase = lane * 17;          // PAD(lane*16): stride 17 -> conflict-free
  float rw[16], rh[16];
  for (int i = 0; i < 16; ++i) { rw[i] = wbuf[pbase + i]; rh[i] = hbuf[pbase + i]; }
  float mw = -1e30f, mh = -1e30f;
  for (int i = 0; i < 16; ++i) { mw = fmaxf(mw, rw[i]); mh = fmaxf(mh, rh[i]); }
  mw = wred_max(mw); mh = wred_max(mh);
  float sw_ = 0.f, sh_ = 0.f;
  for (int i = 0; i < 16; ++i) { sw_ += expf(rw[i] - mw); sh_ += expf(rh[i] - mh); }
  sw_ = wred_sum(sw_); sh_ = wred_sum(sh_);
  const float CW = 1.0f - MINW * NBINS;   // 0.5
  const float CH = 1.0f - MINH * NBINS;
  float prw[16], prh[16], lw = 0.f, lh = 0.f;
  for (int i = 0; i < 16; ++i) {
    int j = base + i;
    float wj = (j < NBINS) ? (MINW + CW * expf(rw[i] - mw) / sw_) : 0.0f;
    float hj = (j < NBINS) ? (MINH + CH * expf(rh[i] - mh) / sh_) : 0.0f;
    lw += wj; prw[i] = lw;
    lh += hj; prh[i] = lh;
  }
  const float offw = wscan_excl(lw, lane);
  const float offh = wscan_excl(lh, lane);
  for (int i = 0; i < 16; ++i) { prw[i] += offw; prh[i] += offh; }

  // coordinate targets + bin search (count of prefix sums <= T)
  const float ab = WINB - WINA;
  const float x0 = ((coords[fid * 2 + 0] - WINA) / ab - 0.5f) * 2.0f;
  const float x1 = ((coords[fid * 2 + 1] - WINA) / ab - 0.5f) * 2.0f;
  const bool in0 = (x0 >= -1.0f) && (x0 <= 1.0f);
  const bool in1 = (x1 >= -1.0f) && (x1 <= 1.0f);
  const float yc0 = fminf(fmaxf(x0, -1.0f), 1.0f);
  const float yc1 = fminf(fmaxf(x1, -1.0f), 1.0f);
  const float T0 = (yc0 + 1.0f) * 0.5f, T1 = (yc1 + 1.0f) * 0.5f;
  int c0 = 0, c1 = 0;
  for (int i = 0; i < 16; ++i) {
    int j = base + i;
    if (j <= NBINS - 2) {
      if (prh[i] <= T0) c0++;
      if (prh[i] <= T1) c1++;
    }
  }
  c0 = wred_sumi(c0); c1 = wred_sumi(c1);
  const int idx0 = min(c0 + ((yc0 >= 1.0f) ? 1 : 0), NBINS - 1);
  const int idx1 = min(c1 + ((yc1 >= 1.0f) ? 1 : 0), NBINS - 1);

  // publish inclusive prefixes so any lane can read arbitrary bins
  for (int i = 0; i < 16; ++i) { wbuf[pbase + i] = prw[i]; hbuf[pbase + i] = prh[i]; }
  lds_fence();

  float total = 0.0f;
  for (int co = 0; co < 2; ++co) {
    const int idx = co ? idx1 : idx0;
    const float yc = co ? yc1 : yc0;
    const bool inside = co ? in1 : in0;
    const float Hm1 = (idx == 0) ? 0.0f : hbuf[PAD(idx - 1)];
    const float Hm  = hbuf[PAD(idx)];
    const float Wm1 = (idx == 0) ? 0.0f : wbuf[PAD(idx - 1)];
    const float Wm  = wbuf[PAD(idx)];
    const float ch_k  = (idx == 0) ? -1.0f : 2.0f * Hm1 - 1.0f;
    const float ch_k1 = (idx == NBINS - 1) ? 1.0f : 2.0f * Hm - 1.0f;
    const float hk = ch_k1 - ch_k;
    const float cw_k  = (idx == 0) ? -1.0f : 2.0f * Wm1 - 1.0f;
    const float cw_k1 = (idx == NBINS - 1) ? 1.0f : 2.0f * Wm - 1.0f;
    const float wk = cw_k1 - cw_k;
    const float dk  = (idx == 0)         ? 1.0f : dval(swg, mixg, hv, lane, 2 * NBINS + idx - 1);
    const float dk1 = (idx == NBINS - 1) ? 1.0f : dval(swg, mixg, hv, lane, 2 * NBINS + idx);
    const float delta = hk / wk;
    const float s2 = dk + dk1 - 2.0f * delta;
    const float dy = yc - ch_k;
    const float qa = dy * s2 + hk * (delta - dk);
    const float qb = hk * dk - dy * s2;
    const float qc = -delta * dy;
    const float disc = qb * qb - 4.0f * qa * qc;
    const float root = 2.0f * qc / (-qb - sqrtf(disc));
    const float th1m = root * (1.0f - root);
    const float den = delta + s2 * th1m;
    const float om = 1.0f - root;
    const float num = delta * delta * (dk1 * root * root + 2.0f * delta * th1m + dk * om * om);
    const float lad = logf(num) - 2.0f * logf(den);
    if (inside) total -= lad;           // -logabsdet, zero outside
  }
  if (lane == 0) {
    float contrib = -(2.0f * (logf(0.5f) - logf(ab)) + total);
    atomicAdd(out, contrib);
  }
}

// ---------- kernel 4: rho GEMM (WMMA) fused with Poisson count log-likelihood ----------
// grid: 19 n-tiles; block: 256 (8 waves = 8 m-tiles covering 128 cells)
__global__ void rho_kernel(const float* __restrict__ h, const float* __restrict__ rho_w,
                           const float* __restrict__ rho_bias, const float* __restrict__ libsize,
                           const int* __restrict__ genes_oi, const int* __restrict__ cells_oi,
                           const float* __restrict__ counts, float* __restrict__ out) {
  const int lane = threadIdx.x & 31, wv = threadIdx.x >> 5;
  const int khalf = lane >> 4;
  const int m = wv * 16 + (lane & 15);
  const int n = blockIdx.x * 16 + (lane & 15);
  const bool nvalid = (n < NGOI);
  const int g = nvalid ? genes_oi[n] : 0;
  v16h a, b; v8f cacc = {};
  for (int v = 0; v < 8; ++v) {
    int kb = ((v < 4) ? 0 : 16) + khalf * 8 + (v & 3) * 2;
    a[2 * v]     = (_Float16)h[m * NHID + kb];
    a[2 * v + 1] = (_Float16)h[m * NHID + kb + 1];
  }
  for (int v = 0; v < 8; ++v) {
    int kb = ((v < 4) ? 0 : 16) + khalf * 8 + (v & 3) * 2;
    b[2 * v]     = nvalid ? (_Float16)rho_w[g * NHID + kb]     : (_Float16)0.0f;
    b[2 * v + 1] = nvalid ? (_Float16)rho_w[g * NHID + kb + 1] : (_Float16)0.0f;
  }
  cacc = __builtin_amdgcn_wmma_f32_16x16x32_f16(false, a, false, b,
                                                (short)0, cacc, false, false);
  float acc = 0.0f;
  if (nvalid) {
    const float rb = rho_bias[g];
    for (int r = 0; r < 8; ++r) {
      const int mm = wv * 16 + r + khalf * 8;
      const float rate = rb * expf(cacc[r]) * libsize[cells_oi[mm]];
      const float cnt = counts[mm * NGOI + n];
      acc += cnt * logf(rate) - rate - lgammaf(cnt + 1.0f);
    }
  }
  acc = wred_sum(acc);
  if (lane == 0) atomicAdd(out, -acc);
}

extern "C" void kernel_launch(void* const* d_in, const int* in_sizes, int n_in,
                              void* d_out, int out_size, void* d_ws, size_t ws_size,
                              hipStream_t stream) {
  const float* latent   = (const float*)d_in[0];
  const float* coords   = (const float*)d_in[1];
  const float* W1       = (const float*)d_in[2];
  const float* b1       = (const float*)d_in[3];
  const float* g1       = (const float*)d_in[4];
  const float* be1      = (const float*)d_in[5];
  const float* W2       = (const float*)d_in[6];
  const float* b2       = (const float*)d_in[7];
  const float* g2       = (const float*)d_in[8];
  const float* be2      = (const float*)d_in[9];
  const float* spline_w = (const float*)d_in[10];
  const float* rho_w    = (const float*)d_in[11];
  const float* mix_spl  = (const float*)d_in[12];
  const float* rho_bias = (const float*)d_in[13];
  const float* libsize  = (const float*)d_in[14];
  const int*   genes_oi = (const int*)d_in[15];
  const int*   cells_oi = (const int*)d_in[16];
  const int*   lcxg     = (const int*)d_in[17];
  (void)in_sizes; (void)n_in; (void)out_size; (void)ws_size;

  float* out    = (float*)d_out;
  float* h      = (float*)d_ws;                    // 128*32 floats
  float* counts = h + NCELLS * NHID;               // 128*300 floats

  init_kernel<<<(NCELLS * NGOI + 255) / 256, 256, 0, stream>>>(out, counts);
  mlp_kernel<<<1, 256, 0, stream>>>(latent, W1, b1, g1, be1, W2, b2, g2, be2, h);
  hist_kernel<<<(NFRAG + 255) / 256, 256, 0, stream>>>(lcxg, counts);
  frag_kernel<<<NFRAG / 8, 256, 0, stream>>>(h, coords, spline_w, mix_spl,
                                             genes_oi, lcxg, out);
  rho_kernel<<<(NGOI + 15) / 16, 256, 0, stream>>>(h, rho_w, rho_bias, libsize,
                                                   genes_oi, cells_oi, counts, out);
}
